// LSTM_45140106281361
// MI455X (gfx1250) — matrix operand order repaired and checked by reading, not compiled, not run on hardware
//
#include <hip/hip_runtime.h>

// LSTM: B=64, S=1024, I=H=O=1024, gates z,i,f,o
#define B_  64
#define S_  1024
#define I_  1024
#define H_  1024
#define O_  1024
#define NG_ 4096      // 4*H gate width
#define LDT 40        // LDS tile row stride in bf16 elems (80 B = 16B-aligned; 16DW row + 4DW pad)

#ifndef __has_builtin
#define __has_builtin(x) 0
#endif

#if __has_builtin(__builtin_amdgcn_global_load_async_to_lds_b128)
#define HAVE_ASYNC 1
#else
#define HAVE_ASYNC 0
#endif

#if __has_builtin(__builtin_amdgcn_tensor_load_to_lds)
#define HAVE_TDM 1
#else
#define HAVE_TDM 0
#endif

#if __has_builtin(__builtin_amdgcn_s_wait_asynccnt)
#define WAIT_ASYNC0() __builtin_amdgcn_s_wait_asynccnt(0)
#else
#define WAIT_ASYNC0() asm volatile("s_wait_asynccnt 0" ::: "memory")
#endif

#if __has_builtin(__builtin_amdgcn_s_wait_tensorcnt)
#define WAIT_TENSOR0() __builtin_amdgcn_s_wait_tensorcnt(0)
#else
#define WAIT_TENSOR0() asm volatile("s_wait_tensorcnt 0" ::: "memory")
#endif

#define AS_GLOBAL __attribute__((address_space(1)))
#define AS_LOCAL  __attribute__((address_space(3)))

typedef __attribute__((ext_vector_type(16))) __bf16 v16bf;
typedef __attribute__((ext_vector_type(8)))  __bf16 v8bf;
typedef __attribute__((ext_vector_type(8)))  float  v8f;

#if HAVE_ASYNC
// Builtin prototype (from clang diagnostic): pointee is GCC-style 'int __vector(4)',
// global operand in AS(1) (printed '__device__'), LDS operand in AS(3).
typedef int v4i_a __attribute__((vector_size(4 * sizeof(int))));

// 16B async copy global -> LDS (per-lane VDST = LDS byte offset; ASYNCcnt tracked).
// AS(3) pointers are 32-bit: inttoptr of the generic address truncates to the LDS
// byte offset, matching the ISA, which takes the LDS address from the low 32 bits.
static __device__ inline void async_copy_b128(void* lds_dst, const void* gsrc) {
  __builtin_amdgcn_global_load_async_to_lds_b128(
      (AS_GLOBAL v4i_a*)(uintptr_t)gsrc,
      (AS_LOCAL  v4i_a*)(uintptr_t)lds_dst, 0, 0);
}
#endif

#if HAVE_TDM
typedef __attribute__((ext_vector_type(4))) unsigned int u32x4;
typedef __attribute__((ext_vector_type(8))) int i32x8;
typedef __attribute__((ext_vector_type(4))) int i32x4;

static __device__ inline void tdm_raw(u32x4 g0, i32x8 g1, i32x4 g2, i32x4 g3) {
#if __clang_major__ >= 23
  i32x8 gx = {};
  __builtin_amdgcn_tensor_load_to_lds(g0, g1, g2, g3, gx, 0);
#else
  __builtin_amdgcn_tensor_load_to_lds(g0, g1, g2, g3, 0);
#endif
}

// 2D/3D bf16 tile DMA: tile tl0(K-elems) x tl1(rows) x tl2(planes), element strides s0/s1.
// LDS rows padded by TDM: pad_interval=3 (16 DWORDs = 32 bf16), pad_amount=3 (4 DWORDs = 8 bf16)
// -> effective LDS row stride = 40 bf16 = LDT.
static __device__ inline void tdm_load_tile_bf16(
    unsigned lds_addr, const void* gptr,
    unsigned tl0, unsigned tl1, unsigned tl2,
    unsigned long long s0, unsigned long long s1) {
  const unsigned long long ga = (unsigned long long)(uintptr_t)gptr;
  const unsigned td0 = 1u << 20, td1 = 1u << 20, td2 = 16;   // generous extents (tiles stay in range)
  u32x4 g0;
  g0[0] = 1u;                                               // count=1, user descriptor
  g0[1] = lds_addr;                                         // LDS byte address
  g0[2] = (unsigned)ga;                                     // global_addr[31:0]
  g0[3] = (unsigned)((ga >> 32) & 0x01ffffffu) | (2u << 30);// global_addr[56:32] | type=2
  i32x8 g1;
  g1[0] = (int)((1u << 16) | (1u << 20) | (3u << 22) | (3u << 25)); // data_size=2B, pad_en, 16DW/4DW
  g1[1] = (int)((td0 & 0xffffu) << 16);                     // [47:32]=0 barrier addr, [63:48]=td0.lo
  g1[2] = (int)((td0 >> 16) | ((td1 & 0xffffu) << 16));     // td0.hi, td1.lo
  g1[3] = (int)((td1 >> 16) | (tl0 << 16));                 // td1.hi, tile_dim0
  g1[4] = (int)(tl1 | (tl2 << 16));                         // tile_dim1, tile_dim2
  g1[5] = (int)(unsigned)s0;                                // dim0_stride[31:0]
  g1[6] = (int)(((unsigned)(s0 >> 32) & 0xffffu) |
                (((unsigned)s1 & 0xffffu) << 16));          // dim0_stride[47:32], dim1_stride[15:0]
  g1[7] = (int)(unsigned)(s1 >> 16);                        // dim1_stride[47:16]
  i32x4 g2; g2[0] = (int)td2; g2[1] = 0; g2[2] = 0; g2[3] = 0;
  i32x4 g3; g3[0] = 0; g3[1] = 0; g3[2] = 0; g3[3] = 0;
  tdm_raw(g0, g1, g2, g3);
}
#endif

// ---- WMMA fragment loaders (wave32, 16x16x32 bf16 layouts per CDNA5 ISA 7.12.2) ----
static __device__ inline v16bf lds_load_a_frag(const __bf16* base, int row0, int lane) {
  const int r  = row0 + (lane & 15);
  const int kh = (lane >> 4) * 8;
  const __bf16* p = base + r * LDT + kh;
  v16bf f;
  *(v8bf*)&f       = *(const v8bf*)(p);
  *((v8bf*)&f + 1) = *(const v8bf*)(p + 16);
  return f;
}
static __device__ inline v16bf lds_load_b_frag(const __bf16* base, int col0, int lane) {
  const int c  = col0 + (lane & 15);
  const int kh = (lane >> 4) * 8;
  const __bf16* p = base + c * LDT + kh;
  v16bf f;
  *(v8bf*)&f       = *(const v8bf*)(p);
  *((v8bf*)&f + 1) = *(const v8bf*)(p + 16);
  return f;
}

#define WMMA_BF16(a, b, c) \
  __builtin_amdgcn_wmma_f32_16x16x32_bf16(false, (a), false, (b), (short)0, (c), false, false)

// ---- fp32 -> bf16 weight conversion ----
__global__ void cvt_f32_bf16_kernel(const float* __restrict__ src, __bf16* __restrict__ dst, long n) {
  long i = (long)blockIdx.x * blockDim.x + threadIdx.x;
  long stride = (long)gridDim.x * blockDim.x;
  for (; i < n; i += stride) dst[i] = (__bf16)src[i];
}

// ---- h0 = 0 (bf16), c0 = 0 (fp32) ----
__global__ void init_state_kernel(__bf16* __restrict__ h, float* __restrict__ c, int n) {
  int i = blockIdx.x * blockDim.x + threadIdx.x;
  if (i < n) { h[i] = (__bf16)0.0f; c[i] = 0.0f; }
}

// ==== Phase 1: xp[s][b][n] = sum_i x[b][s][i] * Wx[n][i] + bx[n] + bh[n]  (bf16 out, f32 accum) ====
__global__ __launch_bounds__(256) void gemm_xp_kernel(
    const float* __restrict__ X, const __bf16* __restrict__ W,
    const float* __restrict__ bx, const float* __restrict__ bh,
    __bf16* __restrict__ xp) {
  __shared__ __align__(16) __bf16 aT[64 * LDT];
  __shared__ __align__(16) __bf16 bT[128 * LDT];

  const int tid  = threadIdx.x;
  const int lane = tid & 31;
  const int w    = tid >> 5;
  const int wm   = w >> 2;
  const int wn   = w & 3;
  const int s_idx = blockIdx.y;
  const int n0    = blockIdx.x * 128;

  v8f acc[2][2] = {};

  const int ar = tid >> 2, aseg = (tid & 3) * 8;
  const float* gA = X + ((size_t)ar * S_ + s_idx) * I_ + aseg;

  for (int kk = 0; kk < I_; kk += 32) {
#if HAVE_TDM
    // B tile 128x32 bf16 via Tensor Data Mover (wave 0 issues; pad gives LDT stride)
    if (w == 0)
      tdm_load_tile_bf16((unsigned)(uintptr_t)(void*)bT,
                         W + (size_t)n0 * I_ + kk,
                         /*tl0=*/32, /*tl1=*/128, /*tl2=*/0,
                         /*s0=*/(unsigned long long)I_, /*s1=*/0ull);
#else
    #pragma unroll
    for (int j = 0; j < 2; ++j) {
      int segid = tid * 2 + j;
      int br = segid >> 2, bseg = (segid & 3) * 8;
      *(v8bf*)&bT[br * LDT + bseg] =
          *(const v8bf*)(W + (size_t)(n0 + br) * I_ + kk + bseg);
    }
#endif
    // A tile (fp32 -> bf16 on the fly; needs VALU, so stays on the VGPR path)
    {
      const float* p = gA + kk;
      __bf16* d = &aT[ar * LDT + aseg];
      #pragma unroll
      for (int e = 0; e < 8; ++e) d[e] = (__bf16)p[e];
      if (kk + 32 < I_) __builtin_prefetch(gA + kk + 32, 0, 0);
    }
#if HAVE_TDM
    if (w == 0) WAIT_TENSOR0();
#endif
    __syncthreads();

    v16bf afr[2], bfr[2];
    #pragma unroll
    for (int i = 0; i < 2; ++i) afr[i] = lds_load_a_frag(aT, wm * 32 + i * 16, lane);
    #pragma unroll
    for (int j = 0; j < 2; ++j) bfr[j] = lds_load_b_frag(bT, wn * 32 + j * 16, lane);
    #pragma unroll
    for (int i = 0; i < 2; ++i)
      #pragma unroll
      for (int j = 0; j < 2; ++j)
        acc[i][j] = WMMA_BF16(afr[i], bfr[j], acc[i][j]);
    __syncthreads();
  }

  const int mhi  = ((lane >> 4) & 1) * 8;
  const int ncol = lane & 15;
  #pragma unroll
  for (int i = 0; i < 2; ++i)
    #pragma unroll
    for (int j = 0; j < 2; ++j)
      #pragma unroll
      for (int v = 0; v < 8; ++v) {
        int mm = wm * 32 + i * 16 + v + mhi;
        int nn = n0 + wn * 32 + j * 16 + ncol;
        float val = acc[i][j][v] + bx[nn] + bh[nn];
        xp[((size_t)s_idx * B_ + mm) * NG_ + nn] = (__bf16)val;
      }
}

// ==== Phase 2: one recurrent step.  g = xp_t + h_in @ Rh^T ; gates ; update c, h_out ====
__global__ __launch_bounds__(256) void lstm_step_kernel(
    const __bf16* __restrict__ xp_t, const __bf16* __restrict__ Rh,
    const __bf16* __restrict__ h_in, __bf16* __restrict__ h_out,
    float* __restrict__ c) {
  __shared__ __align__(16) __bf16 aT[64 * LDT];
  __shared__ __align__(16) __bf16 bT[128 * LDT];
  __shared__ float gT[4][64][32];

  const int tid  = threadIdx.x;
  const int lane = tid & 31;
  const int w    = tid >> 5;
  const int gate = w >> 1;
  const int half = w & 1;
  const int hc0  = blockIdx.x * 32;

  v8f acc[4] = {};
  const int ar = tid >> 2, aseg = (tid & 3) * 8;

  for (int kk = 0; kk < H_; kk += 32) {
#if HAVE_TDM
    // Rh tile as one 3D DMA: 32 K-elems x 32 rows x 4 gates, strides H and H*H
    if (w == 0)
      tdm_load_tile_bf16((unsigned)(uintptr_t)(void*)bT,
                         Rh + (size_t)hc0 * H_ + kk,
                         /*tl0=*/32, /*tl1=*/32, /*tl2=*/4,
                         /*s0=*/(unsigned long long)H_,
                         /*s1=*/(unsigned long long)H_ * H_);
#else
    #pragma unroll
    for (int j = 0; j < 2; ++j) {
      int segid = tid * 2 + j;
      int br = segid >> 2, bseg = (segid & 3) * 8;
      int nG = (br >> 5) * H_ + hc0 + (br & 31);
      *(v8bf*)&bT[br * LDT + bseg] =
          *(const v8bf*)(Rh + (size_t)nG * H_ + kk + bseg);
    }
#endif
#if HAVE_ASYNC
    async_copy_b128(&aT[ar * LDT + aseg], h_in + (size_t)ar * H_ + kk + aseg);
    WAIT_ASYNC0();
#else
    *(v8bf*)&aT[ar * LDT + aseg] =
        *(const v8bf*)(h_in + (size_t)ar * H_ + kk + aseg);
#endif
#if HAVE_TDM
    if (w == 0) WAIT_TENSOR0();
#endif
    __syncthreads();

    v16bf bfr = lds_load_b_frag(bT, gate * 32 + half * 16, lane);
    #pragma unroll
    for (int i = 0; i < 4; ++i) {
      v16bf afr = lds_load_a_frag(aT, i * 16, lane);
      acc[i] = WMMA_BF16(afr, bfr, acc[i]);
    }
    __syncthreads();
  }

  const int mhi  = ((lane >> 4) & 1) * 8;
  const int ncol = lane & 15;
  const int col  = half * 16 + ncol;
  const int nG   = gate * H_ + hc0 + col;
  #pragma unroll
  for (int i = 0; i < 4; ++i)
    #pragma unroll
    for (int v = 0; v < 8; ++v) {
      int m = i * 16 + v + mhi;
      gT[gate][m][col] = acc[i][v] + (float)xp_t[(size_t)m * NG_ + nG];
    }
  __syncthreads();

  for (int idx = tid; idx < 64 * 32; idx += 256) {
    int m = idx >> 5, cc = idx & 31;
    int hcol = hc0 + cc;
    float z  = tanhf(gT[0][m][cc]);
    float ig = 1.0f / (1.0f + expf(-gT[1][m][cc]));
    float fg = 1.0f / (1.0f + expf(-gT[2][m][cc]));
    float og = 1.0f / (1.0f + expf(-gT[3][m][cc]));
    size_t off = (size_t)m * H_ + hcol;
    float cn = fg * c[off] + ig * z;
    float hn = og * tanhf(cn);
    c[off] = cn;
    h_out[off] = (__bf16)hn;
  }
}

// ==== Phase 3: out = h_final @ Wout^T + bout  (M=64, N=1024, K=1024, fp32 out) ====
__global__ __launch_bounds__(256) void gemm_out_kernel(
    const __bf16* __restrict__ Hf, const __bf16* __restrict__ Wo,
    const float* __restrict__ bout, float* __restrict__ out) {
  __shared__ __align__(16) __bf16 aT[64 * LDT];
  __shared__ __align__(16) __bf16 bT[128 * LDT];

  const int tid  = threadIdx.x;
  const int lane = tid & 31;
  const int w    = tid >> 5;
  const int wm   = w >> 2;
  const int wn   = w & 3;
  const int n0   = blockIdx.x * 128;

  v8f acc[2][2] = {};
  const int ar = tid >> 2, aseg = (tid & 3) * 8;

  for (int kk = 0; kk < H_; kk += 32) {
#if HAVE_TDM
    if (w == 0)
      tdm_load_tile_bf16((unsigned)(uintptr_t)(void*)bT,
                         Wo + (size_t)n0 * H_ + kk,
                         32, 128, 0, (unsigned long long)H_, 0ull);
#else
    #pragma unroll
    for (int j = 0; j < 2; ++j) {
      int segid = tid * 2 + j;
      int br = segid >> 2, bseg = (segid & 3) * 8;
      *(v8bf*)&bT[br * LDT + bseg] =
          *(const v8bf*)(Wo + (size_t)(n0 + br) * H_ + kk + bseg);
    }
#endif
#if HAVE_ASYNC
    async_copy_b128(&aT[ar * LDT + aseg], Hf + (size_t)ar * H_ + kk + aseg);
    WAIT_ASYNC0();
#else
    *(v8bf*)&aT[ar * LDT + aseg] =
        *(const v8bf*)(Hf + (size_t)ar * H_ + kk + aseg);
#endif
#if HAVE_TDM
    if (w == 0) WAIT_TENSOR0();
#endif
    __syncthreads();

    v16bf afr[2], bfr[2];
    #pragma unroll
    for (int i = 0; i < 2; ++i) afr[i] = lds_load_a_frag(aT, wm * 32 + i * 16, lane);
    #pragma unroll
    for (int j = 0; j < 2; ++j) bfr[j] = lds_load_b_frag(bT, wn * 32 + j * 16, lane);
    #pragma unroll
    for (int i = 0; i < 2; ++i)
      #pragma unroll
      for (int j = 0; j < 2; ++j)
        acc[i][j] = WMMA_BF16(afr[i], bfr[j], acc[i][j]);
    __syncthreads();
  }

  const int mhi  = ((lane >> 4) & 1) * 8;
  const int ncol = lane & 15;
  #pragma unroll
  for (int i = 0; i < 2; ++i)
    #pragma unroll
    for (int j = 0; j < 2; ++j)
      #pragma unroll
      for (int v = 0; v < 8; ++v) {
        int mm = wm * 32 + i * 16 + v + mhi;
        int nn = n0 + wn * 32 + j * 16 + ncol;
        out[(size_t)mm * O_ + nn] = acc[i][j][v] + bout[nn];
      }
}

extern "C" void kernel_launch(void* const* d_in, const int* in_sizes, int n_in,
                              void* d_out, int out_size, void* d_ws, size_t ws_size,
                              hipStream_t stream) {
  const float* X    = (const float*)d_in[0];  // [B,S,I]
  const float* Wx   = (const float*)d_in[1];  // [4,H,I]
  const float* bx   = (const float*)d_in[2];  // [4,H]
  const float* Rh   = (const float*)d_in[3];  // [4,H,H]
  const float* bh   = (const float*)d_in[4];  // [4,H]
  const float* Wout = (const float*)d_in[5];  // [O,H]
  const float* bout = (const float*)d_in[6];  // [O]
  float* out = (float*)d_out;                 // [B,O]

  char* p = (char*)d_ws;
  auto carve = [&](size_t bytes) {
    void* r = (void*)p;
    p += (bytes + 255) & ~(size_t)255;
    return r;
  };
  __bf16* xp     = (__bf16*)carve((size_t)S_ * B_ * NG_ * sizeof(__bf16)); // 512 MB
  __bf16* Wxbf   = (__bf16*)carve((size_t)NG_ * I_ * sizeof(__bf16));      // 8 MB
  __bf16* Rhbf   = (__bf16*)carve((size_t)NG_ * H_ * sizeof(__bf16));      // 8 MB
  __bf16* Wobf   = (__bf16*)carve((size_t)O_ * H_ * sizeof(__bf16));       // 2 MB
  __bf16* hbuf0  = (__bf16*)carve((size_t)B_ * H_ * sizeof(__bf16));
  __bf16* hbuf1  = (__bf16*)carve((size_t)B_ * H_ * sizeof(__bf16));
  float*  cbuf   = (float*) carve((size_t)B_ * H_ * sizeof(float));

  cvt_f32_bf16_kernel<<<4096, 256, 0, stream>>>(Wx,   Wxbf, (long)NG_ * I_);
  cvt_f32_bf16_kernel<<<4096, 256, 0, stream>>>(Rh,   Rhbf, (long)NG_ * H_);
  cvt_f32_bf16_kernel<<<2048, 256, 0, stream>>>(Wout, Wobf, (long)O_ * H_);
  init_state_kernel<<<(B_ * H_ + 255) / 256, 256, 0, stream>>>(hbuf0, cbuf, B_ * H_);

  gemm_xp_kernel<<<dim3(NG_ / 128, S_), 256, 0, stream>>>(X, Wxbf, bx, bh, xp);

  for (int t = 0; t < S_; ++t) {
    const __bf16* hi = (t & 1) ? hbuf1 : hbuf0;
    __bf16*       ho = (t & 1) ? hbuf0 : hbuf1;
    lstm_step_kernel<<<H_ / 32, 256, 0, stream>>>(
        xp + (size_t)t * B_ * NG_, Rhbf, hi, ho, cbuf);
  }

  gemm_out_kernel<<<O_ / 128, 256, 0, stream>>>(hbuf0, Wobf, bout, out);
}
